// Set2SetReadout_27891517620543
// MI455X (gfx1250) — compile-verified, matrix operand order checked by reading, not run on hardware
//
#include <hip/hip_runtime.h>
#include <hip/hip_bf16.h>
#include <math.h>

#define DIM 128
#define SLAB 128                // output columns per GEMM block (LDS weight slab width)
#define LDW 136                 // padded LDS row stride (bf16): 68 dwords -> conflict-free
#define STRIPS 8                // 16-row M-strips per block (one per wave)

typedef __bf16  bf16_t;
typedef __attribute__((ext_vector_type(16))) __bf16 v16bf;
typedef __attribute__((ext_vector_type(8)))  __bf16 v8bf;
typedef __attribute__((ext_vector_type(8)))  float  v8f;

// ---------------------------------------------------------------------------
// CSR segment offsets: segs[g] = lower_bound(batch, g); segs[G] = N.
// batch is sorted, so every graph's nodes are contiguous. No atomics.
// ---------------------------------------------------------------------------
__global__ void seg_bounds_kernel(const int* __restrict__ batch, int* __restrict__ segs,
                                  int Nn, int Gg) {
    int g = blockIdx.x * blockDim.x + threadIdx.x;
    if (g > Gg) return;
    if (g == Gg) { segs[Gg] = Nn; return; }
    int lo = 0, hi = Nn;
    while (lo < hi) {
        int mid = (lo + hi) >> 1;
        if (batch[mid] < g) lo = mid + 1; else hi = mid;
    }
    segs[g] = lo;
}

// ---------------------------------------------------------------------------
// out[g][c] = relu(sum over segment of x[i][c]).  One block (128 thr) per graph,
// thread = feature column -> fully coalesced row reads.
// ---------------------------------------------------------------------------
__global__ void __launch_bounds__(DIM) pool_relu_kernel(const float* __restrict__ x,
                                                        const int* __restrict__ segs,
                                                        float* __restrict__ out) {
    int g = blockIdx.x, c = threadIdx.x;
    int s = segs[g], e = segs[g + 1];
    float acc = 0.f;
    for (int i = s; i < e; ++i) acc += x[(size_t)i * DIM + c];
    out[(size_t)g * DIM + c] = acc > 0.f ? acc : 0.f;
}

// ---------------------------------------------------------------------------
// WMMA GEMM: C[M,Nout] = A[M,128] @ W[Nout,128]^T + bias[Nout]
// REQUIRES: M % 16 == 0, Nout % 128 == 0 (true for all call sites: 128/384).
//  - block: 256 threads = 8 waves. Block covers 128 rows x 128 cols:
//    wave w owns M-strip (blockIdx.x*8 + w)*16 and sweeps all 8 N-tiles of the
//    LDS weight slab -> 32 WMMAs per wave, weight staging amortized 8x.
//  - W slab (128 out-cols x 128 K) in LDS as bf16, padded rows, b128 everywhere
//  - A fragments (all 4 K-steps) preloaded per ISA 16-bit A 16x32 layout:
//    lane (half h = lane>>4, m = lane&15) holds row m, K runs [k0+8h,+8) and
//    [k0+16+8h,+8)  -> float4 global loads + v_cvt_pk_bf16_f32
//  - C/D layout: vgpr r: lanes 0-15 -> (M=r, N=lane); lanes 16-31 -> (M=8+r)
// ---------------------------------------------------------------------------
__global__ void __launch_bounds__(256) gemm_bias_kernel(const float* __restrict__ A,
                                                        const float* __restrict__ W,
                                                        const float* __restrict__ bias,
                                                        float* __restrict__ C,
                                                        int M, int Nout) {
    __shared__ bf16_t sW[SLAB * LDW];

    const int tid = threadIdx.x;
    const int colBase = blockIdx.y * SLAB;

    // stage weight slab -> LDS (f32 -> bf16), 8 elems (one b128) per thread-step
    for (int idx = tid; idx < SLAB * (DIM / 8); idx += 256) {
        int n = idx >> 4;                 // 16 groups of 8 per row
        int k = (idx & 15) << 3;
        const float4* wp = (const float4*)&W[(size_t)(colBase + n) * DIM + k];
        float4 w0 = wp[0];
        float4 w1 = wp[1];
        v8bf v;
        v[0] = (bf16_t)w0.x; v[1] = (bf16_t)w0.y; v[2] = (bf16_t)w0.z; v[3] = (bf16_t)w0.w;
        v[4] = (bf16_t)w1.x; v[5] = (bf16_t)w1.y; v[6] = (bf16_t)w1.z; v[7] = (bf16_t)w1.w;
        *(v8bf*)&sW[n * LDW + k] = v;
    }
    __syncthreads();

    const int wave = tid >> 5;          // 0..7
    const int lane = tid & 31;
    const int half = lane >> 4;         // 0/1
    const int lm   = lane & 15;         // 0..15
    const int tileM = (blockIdx.x * STRIPS + wave) * 16;
    if (tileM >= M) return;             // wave-uniform: EXEC stays all-1s for WMMA

    const float4* __restrict__ arow4 =
        (const float4*)(A + (size_t)(tileM + lm) * DIM);

    // ---- preload all 4 A fragments (K = 0..127) for this strip
    v16bf afrag[4];
#pragma unroll
    for (int kk = 0; kk < 4; ++kk) {
        const int kb = kk * 32 + half * 8;   // multiple of 8
        float4 q0 = arow4[(kb >> 2) + 0];
        float4 q1 = arow4[(kb >> 2) + 1];
        float4 q2 = arow4[(kb >> 2) + 4];    // +16 floats
        float4 q3 = arow4[(kb >> 2) + 5];
        v16bf a;
        a[0]  = (bf16_t)q0.x; a[1]  = (bf16_t)q0.y; a[2]  = (bf16_t)q0.z; a[3]  = (bf16_t)q0.w;
        a[4]  = (bf16_t)q1.x; a[5]  = (bf16_t)q1.y; a[6]  = (bf16_t)q1.z; a[7]  = (bf16_t)q1.w;
        a[8]  = (bf16_t)q2.x; a[9]  = (bf16_t)q2.y; a[10] = (bf16_t)q2.z; a[11] = (bf16_t)q2.w;
        a[12] = (bf16_t)q3.x; a[13] = (bf16_t)q3.y; a[14] = (bf16_t)q3.z; a[15] = (bf16_t)q3.w;
        afrag[kk] = a;
    }

    // ---- sweep the 8 N-tiles of the slab
#pragma unroll
    for (int nt = 0; nt < 8; ++nt) {
        const int nl = nt * 16 + lm;    // output column within slab
        v8f acc = {};
#pragma unroll
        for (int kk = 0; kk < 4; ++kk) {
            const int kb = kk * 32 + half * 8;
            v8bf b0 = *(const v8bf*)&sW[nl * LDW + kb];
            v8bf b1 = *(const v8bf*)&sW[nl * LDW + kb + 16];
            v16bf b;
#pragma unroll
            for (int e = 0; e < 8; ++e) { b[e] = b0[e]; b[8 + e] = b1[e]; }
            acc = __builtin_amdgcn_wmma_f32_16x16x32_bf16(
                /*neg_a=*/false, afrag[kk], /*neg_b=*/false, b,
                /*c_mod=*/(short)0, acc, /*reuse_a=*/false, /*reuse_b=*/false);
        }
        const int colN = colBase + nt * 16 + lm;
        const float bcol = bias[colN];
#pragma unroll
        for (int r = 0; r < 8; ++r) {
            int row = tileM + half * 8 + r;
            C[(size_t)row * Nout + colN] = acc[r] + bcol;
        }
    }
}

// ---------------------------------------------------------------------------
// e[i] = leaky_relu(xl[i] + xr[batch[i]], 0.01) . att      (one wave per node)
// xl is streamed (256 MB/pass) -> nontemporal loads.
// ---------------------------------------------------------------------------
__global__ void __launch_bounds__(256) attn_score_kernel(const float* __restrict__ xl,
                                                         const float* __restrict__ xr,
                                                         const int* __restrict__ batch,
                                                         const float* __restrict__ att,
                                                         float* __restrict__ e, int Nn) {
    int node = blockIdx.x * 8 + (threadIdx.x >> 5);
    int lane = threadIdx.x & 31;
    if (node >= Nn) return;                       // wave-uniform
    int b = batch[node];
    const float* xi = xl + (size_t)node * DIM;
    const float* xg = xr + (size_t)b * DIM;
    float s = 0.f;
#pragma unroll
    for (int j = 0; j < 4; ++j) {
        int c = lane + j * 32;
        float v = __builtin_nontemporal_load(&xi[c]) + xg[c];
        v = v > 0.f ? v : 0.01f * v;
        s += v * att[c];
    }
#pragma unroll
    for (int off = 16; off > 0; off >>= 1) s += __shfl_xor(s, off, 32);
    if (lane == 0) e[node] = s;
}

// ---------------------------------------------------------------------------
// Per-graph softmax stats: m[g] = max(e in segment) (0 if empty, matching the
// isfinite guard), den[g] = sum exp(e - m).
// ---------------------------------------------------------------------------
__global__ void seg_softmax_stats_kernel(const float* __restrict__ e,
                                         const int* __restrict__ segs,
                                         float* __restrict__ m, float* __restrict__ den,
                                         int Gg) {
    int g = blockIdx.x * blockDim.x + threadIdx.x;
    if (g >= Gg) return;
    int s = segs[g], t = segs[g + 1];
    float mx = -INFINITY;
    for (int i = s; i < t; ++i) mx = fmaxf(mx, e[i]);
    if (s >= t) mx = 0.f;
    float d = 0.f;
    for (int i = s; i < t; ++i) d += __expf(e[i] - mx);
    m[g] = mx;
    den[g] = d;
}

// alpha[i] = exp(e[i]-m[b]) / den[b]
__global__ void alpha_kernel(const float* __restrict__ e, const int* __restrict__ batch,
                             const float* __restrict__ m, const float* __restrict__ den,
                             float* __restrict__ alpha, int Nn) {
    int i = blockIdx.x * blockDim.x + threadIdx.x;
    if (i >= Nn) return;
    int b = batch[i];
    alpha[i] = __expf(e[i] - m[b]) / den[b];
}

// h[g][c] = elu( sum_i alpha[i]*xl[i][c] + gat_bias[c] )
__global__ void __launch_bounds__(DIM) aggregate_elu_kernel(const float* __restrict__ xl,
                                                            const float* __restrict__ alpha,
                                                            const int* __restrict__ segs,
                                                            const float* __restrict__ gat_bias,
                                                            float* __restrict__ h) {
    int g = blockIdx.x, c = threadIdx.x;
    int s = segs[g], t = segs[g + 1];
    float acc = 0.f;
    for (int i = s; i < t; ++i)
        acc += alpha[i] * __builtin_nontemporal_load(&xl[(size_t)i * DIM + c]);
    float v = acc + gat_bias[c];
    h[(size_t)g * DIM + c] = v > 0.f ? v : (__expf(v) - 1.f);
}

// out = relu(GRUCell(h->gi, out->gh));  gi/gh are [G, 384] with biases folded in.
__global__ void gru_relu_kernel(const float* __restrict__ gi, const float* __restrict__ gh,
                                float* __restrict__ out, int total) {
    int idx = blockIdx.x * blockDim.x + threadIdx.x;
    if (idx >= total) return;
    int g = idx >> 7, c = idx & (DIM - 1);
    const float* gir = gi + (size_t)g * (3 * DIM);
    const float* ghr = gh + (size_t)g * (3 * DIM);
    float r = 1.f / (1.f + __expf(-(gir[c] + ghr[c])));
    float z = 1.f / (1.f + __expf(-(gir[DIM + c] + ghr[DIM + c])));
    float n = tanhf(gir[2 * DIM + c] + r * ghr[2 * DIM + c]);
    float hp = out[idx];
    float o = (1.f - z) * n + z * hp;
    out[idx] = o > 0.f ? o : 0.f;
}

// ---------------------------------------------------------------------------
extern "C" void kernel_launch(void* const* d_in, const int* in_sizes, int n_in,
                              void* d_out, int out_size, void* d_ws, size_t ws_size,
                              hipStream_t stream) {
    const float* x        = (const float*)d_in[0];
    const int*   batch    = (const int*)  d_in[1];
    const float* Wl       = (const float*)d_in[2];
    const float* bl       = (const float*)d_in[3];
    const float* Wr       = (const float*)d_in[4];
    const float* br       = (const float*)d_in[5];
    const float* att      = (const float*)d_in[6];
    const float* gat_bias = (const float*)d_in[7];
    const float* W_ih     = (const float*)d_in[8];
    const float* W_hh     = (const float*)d_in[9];
    const float* b_ih     = (const float*)d_in[10];
    const float* b_hh     = (const float*)d_in[11];
    const float* Wfc      = (const float*)d_in[12];
    const float* bfc      = (const float*)d_in[13];

    const int Nn = in_sizes[0] / DIM;       // 500000 (divisible by 16)
    const int Gg = out_size / DIM;          // 16384  (divisible by 128)
    float* out = (float*)d_out;

    // workspace carve-out (256B aligned)
    char* w = (char*)d_ws;
    auto alloc = [&](size_t bytes) -> void* {
        void* p = (void*)w;
        w += (bytes + 255) & ~(size_t)255;
        return p;
    };
    int*   segs  = (int*)  alloc((size_t)(Gg + 1) * sizeof(int));
    float* xl    = (float*)alloc((size_t)Nn * DIM * sizeof(float));
    float* hid   = (float*)alloc((size_t)Gg * DIM * sizeof(float));   // "out" state
    float* xr    = (float*)alloc((size_t)Gg * DIM * sizeof(float));
    float* hbuf  = (float*)alloc((size_t)Gg * DIM * sizeof(float));
    float* ev    = (float*)alloc((size_t)Nn * sizeof(float));
    float* alpha = (float*)alloc((size_t)Nn * sizeof(float));
    float* mbuf  = (float*)alloc((size_t)Gg * sizeof(float));
    float* den   = (float*)alloc((size_t)Gg * sizeof(float));
    float* gi    = (float*)alloc((size_t)Gg * 3 * DIM * sizeof(float));
    float* gh    = (float*)alloc((size_t)Gg * 3 * DIM * sizeof(float));
    (void)ws_size; (void)n_in;

    const int stripsN = Nn / 16;            // 31250
    const int stripsG = Gg / 16;            // 1024

    // 1. CSR segment offsets from sorted batch
    seg_bounds_kernel<<<(Gg + 256) / 256, 256, 0, stream>>>(batch, segs, Nn, Gg);

    // 2. out0 = relu(segment_sum(x))
    pool_relu_kernel<<<Gg, DIM, 0, stream>>>(x, segs, hid);

    // 3. xl = x @ Wl^T + bl   (the big streaming WMMA GEMM)
    {
        dim3 grid((stripsN + STRIPS - 1) / STRIPS, 1);
        gemm_bias_kernel<<<grid, 256, 0, stream>>>(x, Wl, bl, xl, Nn, DIM);
    }

    // 4. T = 3 message-passing / GRU steps
    for (int t = 0; t < 3; ++t) {
        dim3 gG(stripsG / STRIPS, 1);
        // xr = out @ Wr^T + br
        gemm_bias_kernel<<<gG, 256, 0, stream>>>(hid, Wr, br, xr, Gg, DIM);
        // e = leaky_relu(xl + xr[batch]) . att
        attn_score_kernel<<<(Nn + 7) / 8, 256, 0, stream>>>(xl, xr, batch, att, ev, Nn);
        // segment softmax
        seg_softmax_stats_kernel<<<(Gg + 255) / 256, 256, 0, stream>>>(ev, segs, mbuf, den, Gg);
        alpha_kernel<<<(Nn + 255) / 256, 256, 0, stream>>>(ev, batch, mbuf, den, alpha, Nn);
        // h = elu(segment_sum(alpha * xl) + gat_bias)
        aggregate_elu_kernel<<<Gg, DIM, 0, stream>>>(xl, alpha, segs, gat_bias, hbuf);
        // GRU gates: gi = h @ W_ih^T + b_ih ; gh = out @ W_hh^T + b_hh   (Nout = 384)
        dim3 gGRU(stripsG / STRIPS, 3);
        gemm_bias_kernel<<<gGRU, 256, 0, stream>>>(hbuf, W_ih, b_ih, gi, Gg, 3 * DIM);
        gemm_bias_kernel<<<gGRU, 256, 0, stream>>>(hid,  W_hh, b_hh, gh, Gg, 3 * DIM);
        // out = relu(gru(...)) in place
        gru_relu_kernel<<<(Gg * DIM + 255) / 256, 256, 0, stream>>>(gi, gh, hid, Gg * DIM);
    }

    // 5. d_out = out @ Wfc^T + bfc
    {
        dim3 grid(stripsG / STRIPS, 1);
        gemm_bias_kernel<<<grid, 256, 0, stream>>>(hid, Wfc, bfc, out, Gg, DIM);
    }
}
